// LearnablePQ_57415122813094
// MI455X (gfx1250) — compile-verified, matrix operand order checked by reading, not run on hardware
//
#include <hip/hip_runtime.h>
#include <hip/hip_bf16.h>
#include <math.h>
#include <stdint.h>

typedef __bf16 bf16_t;
typedef bf16_t v16bf __attribute__((ext_vector_type(16)));
typedef float  v8f   __attribute__((ext_vector_type(8)));
typedef unsigned int v4u __attribute__((ext_vector_type(4)));
typedef int v8i __attribute__((ext_vector_type(8)));
typedef int v4i __attribute__((ext_vector_type(4)));

#define IN_DIM 768
#define H1D 512
#define H2D 384
#define TGT 256
#define MT 64
#define THREADS 512   // 16 waves: 4 M-stripes x 4 N-quarters

#if defined(__has_builtin)
# if __has_builtin(__builtin_amdgcn_tensor_load_to_lds) && __has_builtin(__builtin_amdgcn_s_wait_tensorcnt)
#  define HAVE_TDM 1
# endif
#endif
#ifndef HAVE_TDM
# define HAVE_TDM 0
#endif

union FragU { uint4 q[2]; v16bf v; };
union Half4 { bf16_t h[4]; uint2 q; };

__device__ __forceinline__ float gelu_exact(float v) {
    return 0.5f * v * (1.0f + erff(v * 0.70710678118654752f));
}

__device__ __forceinline__ v16bf frag_ld(const bf16_t* base, int lo, int hi) {
    FragU f;
    f.q[0] = *(const uint4*)(base + lo);
    f.q[1] = *(const uint4*)(base + hi);
    return f.v;
}

#if HAVE_TDM
// Issue a 1-D TDM copy: `bytes` (multiple of 8) from global -> LDS.
// D# per CDNA5 ISA 8.3/8.4: count=1, type=2(image), data_size=8B,
// tensor_dim0=tile_dim0=bytes/8, tensor_dim1=1 (tile_dim1..4 unused).
__device__ __forceinline__ void tdm_issue(void* ldsDst, const void* gsrc, int bytes) {
    const unsigned lds = (unsigned)(uintptr_t)ldsDst;       // low 32b = LDS offset
    const unsigned long long ga = (unsigned long long)(uintptr_t)gsrc;
    const unsigned n8 = ((unsigned)bytes) >> 3;
    v4u g0 = { 1u,                                          // count=1, user mode
               lds,                                         // lds_addr
               (unsigned)(ga & 0xFFFFFFFFu),                // global_addr[31:0]
               (unsigned)((ga >> 32) & 0x01FFFFFFu) | (2u << 30) }; // [56:32] | type=2
    v8i g1 = { (int)(3u << 16),                             // data_size=8B
               (int)((n8 & 0xFFFFu) << 16),                 // tensor_dim0[15:0]
               (int)(((n8 >> 16) & 0xFFFFu) | (1u << 16)),  // tensor_dim0[31:16], tensor_dim1=1
               (int)((n8 & 0xFFFFu) << 16),                 // tile_dim0 = n8
               0, 0, 0, 0 };
    v4i z4 = { 0, 0, 0, 0 };
    v8i z8 = { 0, 0, 0, 0, 0, 0, 0, 0 };
    __builtin_amdgcn_tensor_load_to_lds(g0, g1, z4, z4, z8, 0);
}
#endif

// One K=32 chunk of WMMA for this wave, B-fragment rotation for overlap.
template<int NTH>
__device__ __forceinline__ void gemm_chunk(const bf16_t* aBase, int aStride, int kbase,
                                           const bf16_t* wst, v8f* acc,
                                           int stripe, int nq, int lane) {
    const int arow  = stripe * 16 + (lane & 15);
    const int khalf = lane >> 4;
    const int lo = arow * aStride + kbase + khalf * 8;
    v16bf a = frag_ld(aBase, lo, lo + 16);
    const int base = (nq * NTH) * 512 + lane * 16;     // 512 bf16 per 16x32 tile
    v16bf bc = frag_ld(wst, base, base + 8);
#pragma unroll
    for (int j = 0; j < NTH; ++j) {
        v16bf bn;
        if (j + 1 < NTH) {
            const int bo = base + (j + 1) * 512;
            bn = frag_ld(wst, bo, bo + 8);
        }
        acc[j] = __builtin_amdgcn_wmma_f32_16x16x32_bf16(
            false, a, false, bc, (short)0, acc[j], false, false);
        if (j + 1 < NTH) bc = bn;
    }
}

// Weight-only GEMM phase with double-buffered TDM (or fallback copy) staging.
template<int NTH, int NCH>
__device__ __forceinline__ void gemm_pipe(const bf16_t* aBase, int aStride,
                                          const bf16_t* __restrict__ wsrc, int chunkBytes,
                                          bf16_t* wb0, bf16_t* wb1, v8f* acc,
                                          int stripe, int nq, int lane, int wave, int tid) {
    const int chunkElems = chunkBytes / 2;
#if HAVE_TDM
    if (wave == 0) tdm_issue(wb0, wsrc, chunkBytes);
#else
    { const uint4* s = (const uint4*)wsrc; uint4* d = (uint4*)wb0;
      for (int i = tid; i < chunkBytes / 16; i += THREADS) d[i] = s[i]; }
#endif
    for (int ch = 0; ch < NCH; ++ch) {
#if HAVE_TDM
        if (wave == 0) __builtin_amdgcn_s_wait_tensorcnt(0);
#endif
        __syncthreads();
        if (ch + 1 < NCH) {
            bf16_t* nb = ((ch + 1) & 1) ? wb1 : wb0;
            const bf16_t* ns = wsrc + (size_t)(ch + 1) * chunkElems;
#if HAVE_TDM
            if (wave == 0) tdm_issue(nb, ns, chunkBytes);
#else
            { const uint4* s = (const uint4*)ns; uint4* d = (uint4*)nb;
              for (int i = tid; i < chunkBytes / 16; i += THREADS) d[i] = s[i]; }
#endif
        }
        gemm_chunk<NTH>(aBase, aStride, ch * 32, (ch & 1) ? wb1 : wb0,
                        acc, stripe, nq, lane);
    }
}

// ---------- pre-pack kernels ----------
__global__ void pack_w_kernel(const float* __restrict__ W, bf16_t* __restrict__ out,
                              int K, int N) {
    int tid = blockIdx.x * blockDim.x + threadIdx.x;
    const int NT = N / 16;
    const int total = (K / 32) * NT * 64;
    if (tid >= total) return;
    const int half  = tid & 1;
    const int l     = (tid >> 1) & 31;
    const int ct    = tid >> 6;
    const int t     = ct % NT;
    const int chunk = ct / NT;
    const int n     = t * 16 + (l & 15);
    const int khalf = l >> 4;
    const int k0    = chunk * 32 + half * 16 + khalf * 8;
#pragma unroll
    for (int j = 0; j < 8; ++j)
        out[tid * 8 + j] = (bf16_t)W[(size_t)(k0 + j) * N + n];
}

__global__ void pack_cb_kernel(const float* __restrict__ cb, bf16_t* __restrict__ out) {
    int tid = blockIdx.x * blockDim.x + threadIdx.x;
    if (tid >= 16 * 16 * 64) return;
    const int half = tid & 1;
    const int l    = (tid >> 1) & 31;
    const int st   = tid >> 6;
    const int t    = st % 16;
    const int sub  = st / 16;
    const int n    = t * 16 + (l & 15);
    const int khalf = l >> 4;
#pragma unroll
    for (int j = 0; j < 8; ++j) {
        bf16_t v = (bf16_t)0.0f;
        if (half == 0) {
            const int d = khalf * 8 + j;
            v = (bf16_t)cb[((size_t)sub * 256 + n) * 16 + d];
        }
        out[tid * 8 + j] = v;
    }
}

__global__ void c2_kernel(const float* __restrict__ cb, float* __restrict__ c2) {
    int tid = blockIdx.x * blockDim.x + threadIdx.x;
    if (tid >= 16 * 256) return;
    const float* p = cb + (size_t)tid * 16;
    float s = 0.0f;
#pragma unroll
    for (int d = 0; d < 16; ++d) s += p[d] * p[d];
    c2[tid] = s;
}

// ---------- fused MLP + PQ kernel ----------
__global__ __launch_bounds__(THREADS) void fused_mlp_pq(
    const float* __restrict__ x,
    const float* __restrict__ b1, const float* __restrict__ b2,
    const float* __restrict__ b3,
    const float* __restrict__ cb,
    const bf16_t* __restrict__ w1p, const bf16_t* __restrict__ w2p,
    const bf16_t* __restrict__ w3p, const bf16_t* __restrict__ cbp,
    const float* __restrict__ c2,
    float* __restrict__ out) {
    __shared__ __align__(16) bf16_t s_h1[MT * H1D];     // 64 KB
    __shared__ __align__(16) bf16_t s_h2[MT * H2D];     // 48 KB
    __shared__ __align__(16) bf16_t s_wstA[32 * 512];   // 32 KB weight buf 0 / encoded
    __shared__ __align__(16) bf16_t s_wstB[32 * 512];   // 32 KB weight buf 1
    __shared__ __align__(16) bf16_t s_astA[MT * 32];    //  4 KB x buf 0
    __shared__ __align__(16) bf16_t s_astB[MT * 32];    //  4 KB x buf 1

    const int tid    = threadIdx.x;
    const int lane   = tid & 31;
    const int wave   = tid >> 5;       // 0..15
    const int stripe = wave & 3;       // M-stripe (16 rows)
    const int nq     = wave >> 2;      // N-quarter
    const int khalf  = lane >> 4;
    const int row0   = blockIdx.x * MT;

    // ================= GEMM1: h1 = gelu(x @ W1 + b1) =================
    v8f acc1[8];
#pragma unroll
    for (int j = 0; j < 8; ++j) acc1[j] = (v8f){0,0,0,0,0,0,0,0};

    const int xrow = tid >> 3, xq = tid & 7;            // 512 thr: 4 floats each
    const float* xbase = x + (size_t)(row0 + xrow) * IN_DIM + xq * 4;
    float4 xr = *(const float4*)xbase;
#if HAVE_TDM
    if (wave == 0) tdm_issue(s_wstA, w1p, 32768);
#else
    { const uint4* s = (const uint4*)w1p; uint4* d = (uint4*)s_wstA;
      for (int i = tid; i < 2048; i += THREADS) d[i] = s[i]; }
#endif
    for (int ch = 0; ch < 24; ++ch) {
        { // commit prefetched x registers -> bf16 staging buffer
            Half4 h4;
            h4.h[0] = (bf16_t)xr.x; h4.h[1] = (bf16_t)xr.y;
            h4.h[2] = (bf16_t)xr.z; h4.h[3] = (bf16_t)xr.w;
            bf16_t* ab = (ch & 1) ? s_astB : s_astA;
            *(uint2*)&ab[xrow * 32 + xq * 4] = h4.q;
        }
#if HAVE_TDM
        if (wave == 0) __builtin_amdgcn_s_wait_tensorcnt(0);
#endif
        __syncthreads();
        if (ch + 1 < 24) {
            xr = *(const float4*)(xbase + (ch + 1) * 32);   // prefetch next x chunk
            bf16_t* nb = ((ch + 1) & 1) ? s_wstB : s_wstA;
            const bf16_t* ns = w1p + (size_t)(ch + 1) * 16384;
#if HAVE_TDM
            if (wave == 0) tdm_issue(nb, ns, 32768);
#else
            { const uint4* s = (const uint4*)ns; uint4* d = (uint4*)nb;
              for (int i = tid; i < 2048; i += THREADS) d[i] = s[i]; }
#endif
        }
        gemm_chunk<8>((ch & 1) ? s_astB : s_astA, 32, 0,
                      (ch & 1) ? s_wstB : s_wstA, acc1, stripe, nq, lane);
    }
#pragma unroll
    for (int j = 0; j < 8; ++j) {
        const int n = nq * 128 + j * 16 + (lane & 15);
        const float bias = b1[n];
#pragma unroll
        for (int i = 0; i < 8; ++i) {
            const int m = stripe * 16 + khalf * 8 + i;
            s_h1[m * H1D + n] = (bf16_t)gelu_exact(acc1[j][i] + bias);
        }
    }
    __syncthreads();

    // ================= GEMM2: h2 = gelu(h1 @ W2 + b2) =================
    v8f acc2[6];
#pragma unroll
    for (int j = 0; j < 6; ++j) acc2[j] = (v8f){0,0,0,0,0,0,0,0};
    gemm_pipe<6, 16>(s_h1, H1D, w2p, 24576, s_wstA, s_wstB, acc2,
                     stripe, nq, lane, wave, tid);
#pragma unroll
    for (int j = 0; j < 6; ++j) {
        const int n = nq * 96 + j * 16 + (lane & 15);
        const float bias = b2[n];
#pragma unroll
        for (int i = 0; i < 8; ++i) {
            const int m = stripe * 16 + khalf * 8 + i;
            s_h2[m * H2D + n] = (bf16_t)gelu_exact(acc2[j][i] + bias);
        }
    }
    __syncthreads();

    // ================= GEMM3: encoded = h2 @ W3 + b3 =================
    v8f acc3[4];
#pragma unroll
    for (int j = 0; j < 4; ++j) acc3[j] = (v8f){0,0,0,0,0,0,0,0};
    gemm_pipe<4, 12>(s_h2, H2D, w3p, 16384, s_wstA, s_wstB, acc3,
                     stripe, nq, lane, wave, tid);
    __syncthreads();
    bf16_t* s_ebf = s_wstA;                 // reuse: encoded bf16 [64][256]
#pragma unroll
    for (int j = 0; j < 4; ++j) {
        const int n = nq * 64 + j * 16 + (lane & 15);
        const float bias = b3[n];
#pragma unroll
        for (int i = 0; i < 8; ++i) {
            const int m = stripe * 16 + khalf * 8 + i;
            s_ebf[m * TGT + n] = (bf16_t)(acc3[j][i] + bias);
        }
    }
    __syncthreads();

    // ===== PQ: argmin_k (c2[k] - 2 e.c_k) via WMMA + codebook gather =====
    for (int task = wave; task < 64; task += 16) {
        const int st  = task & 3;
        const int sub = task >> 2;
        const int arow = st * 16 + (lane & 15);
        FragU fa;
        fa.q[0] = *(const uint4*)&s_ebf[arow * TGT + sub * 16 + khalf * 8];
        fa.q[1] = make_uint4(0u, 0u, 0u, 0u);    // K 16..31 zero-pad
        const v16bf a = fa.v;

        float bd[8];
        int   bn[8];
#pragma unroll
        for (int i = 0; i < 8; ++i) { bd[i] = 3.0e38f; bn[i] = 0; }

        const int fbase = (sub * 16) * 512 + lane * 16;
        FragU fb;
        fb.q[0] = *(const uint4*)(cbp + fbase);
        fb.q[1] = *(const uint4*)(cbp + fbase + 8);
#pragma unroll
        for (int j = 0; j < 16; ++j) {
            FragU fbn;
            if (j + 1 < 16) {
                const int bo = fbase + (j + 1) * 512;
                fbn.q[0] = *(const uint4*)(cbp + bo);
                fbn.q[1] = *(const uint4*)(cbp + bo + 8);
            }
            v8f c = (v8f){0,0,0,0,0,0,0,0};
            c = __builtin_amdgcn_wmma_f32_16x16x32_bf16(
                false, a, false, fb.v, (short)0, c, false, false);
            const int n = j * 16 + (lane & 15);
            const float c2v = c2[sub * 256 + n];
#pragma unroll
            for (int i = 0; i < 8; ++i) {
                const float d = c2v - 2.0f * c[i];
                if (d < bd[i]) { bd[i] = d; bn[i] = n; }
            }
            if (j + 1 < 16) fb = fbn;
        }
#pragma unroll
        for (int msk = 1; msk <= 8; msk <<= 1) {
#pragma unroll
            for (int i = 0; i < 8; ++i) {
                const float od = __shfl_xor(bd[i], msk, 32);
                const int   on = __shfl_xor(bn[i], msk, 32);
                if (od < bd[i] || (od == bd[i] && on < bn[i])) { bd[i] = od; bn[i] = on; }
            }
        }
#pragma unroll
        for (int i = 0; i < 8; ++i) {
            const int m   = st * 16 + khalf * 8 + i;
            const int col = lane & 15;
            out[(size_t)(row0 + m) * TGT + sub * 16 + col] =
                cb[((size_t)sub * 256 + bn[i]) * 16 + col];
        }
    }
}

// ---------- host launcher ----------
extern "C" void kernel_launch(void* const* d_in, const int* in_sizes, int n_in,
                              void* d_out, int out_size, void* d_ws, size_t ws_size,
                              hipStream_t stream) {
    const float* x  = (const float*)d_in[0];
    const float* W1 = (const float*)d_in[1];
    const float* b1 = (const float*)d_in[2];
    const float* W2 = (const float*)d_in[3];
    const float* b2 = (const float*)d_in[4];
    const float* W3 = (const float*)d_in[5];
    const float* b3 = (const float*)d_in[6];
    const float* cb = (const float*)d_in[7];
    float* out = (float*)d_out;

    char* ws = (char*)d_ws;
    bf16_t* w1p = (bf16_t*)(ws);                   // 786432 B
    bf16_t* w2p = (bf16_t*)(ws + 786432);          // 393216 B
    bf16_t* w3p = (bf16_t*)(ws + 1179648);         // 196608 B
    bf16_t* cbp = (bf16_t*)(ws + 1376256);         // 262144 B
    float*  c2  = (float*)(ws + 1638400);          // 16384 B

    hipLaunchKernelGGL(pack_w_kernel, dim3(192), dim3(256), 0, stream, W1, w1p, 768, 512);
    hipLaunchKernelGGL(pack_w_kernel, dim3(96),  dim3(256), 0, stream, W2, w2p, 512, 384);
    hipLaunchKernelGGL(pack_w_kernel, dim3(48),  dim3(256), 0, stream, W3, w3p, 384, 256);
    hipLaunchKernelGGL(pack_cb_kernel, dim3(64), dim3(256), 0, stream, cb, cbp);
    hipLaunchKernelGGL(c2_kernel, dim3(16), dim3(256), 0, stream, cb, c2);

    hipLaunchKernelGGL(fused_mlp_pq, dim3(65536 / MT), dim3(THREADS), 0, stream,
                       x, b1, b2, b3, cb, w1p, w2p, w3p, cbp, c2, out);
}
